// SpatialGNN_1219770712590
// MI455X (gfx1250) — matrix operand order repaired and checked by reading, not compiled, not run on hardware
//
#include <hip/hip_runtime.h>
#include <hip/hip_fp16.h>

typedef __attribute__((ext_vector_type(16))) _Float16 v16h;
typedef __attribute__((ext_vector_type(2)))  __fp16   v2fp16;
typedef __attribute__((ext_vector_type(8)))  float    v8f;

#define HEADS 4
#define NEG_SLOPE 0.2f
#define LN_EPS 1e-5f

// order-preserving float<->int mapping for atomicMax on floats
__device__ __forceinline__ int f2ord(float f) {
    int b = __float_as_int(f);
    return b ^ ((b >> 31) & 0x7fffffff);
}
__device__ __forceinline__ float ord2f(int o) {
    int b = o ^ ((o >> 31) & 0x7fffffff);
    return __int_as_float(b);
}

// pack two f32 -> packed f16 pair (v_cvt_pk_rtz_f16_f32)
__device__ __forceinline__ unsigned pk2(float a, float b) {
    union { v2fp16 h; unsigned u; } t;
    t.h = __builtin_amdgcn_cvt_pkrtz(a, b);
    return t.u;
}

// ---------------------------------------------------------------------------
// C[M,Nc] = A[M,K] @ B[K,Nc] + bias[Nc]   (f32 in/out, f16 WMMA, f32 accum)
// One 16x16 output tile per wave; WY waves per block share the A tile.
// Requirements: M % 16 == 0, Nc % (16*WY) == 0, K % 2 == 0,
//               and if K % 32 != 0 then K < 32 (single padded chunk).
// LDS tiles hold packed f16 pairs (one uint = 2 consecutive K elements).
// ---------------------------------------------------------------------------
template<int WY, int K>
__launch_bounds__(32 * WY)
__global__ void gemm_wmma(const float* __restrict__ A, const float* __restrict__ B,
                          const float* __restrict__ bias, float* __restrict__ C,
                          int M, int Nc)
{
    static_assert(K % 2 == 0, "K must be even");
    constexpr int NCH = (K + 31) / 32;
    static_assert(K % 32 == 0 || NCH == 1, "partial chunk only when K<32");

    __shared__ unsigned As2[16][17];        // [row][kpair] within chunk
    __shared__ unsigned Bs2[WY][16][17];    // [wave][col][kpair]

    const int tx   = threadIdx.x, ty = threadIdx.y;
    const int tid  = ty * 32 + tx;
    const int m0   = blockIdx.x * 16;
    const int n0   = (blockIdx.y * WY + ty) * 16;
    const int half = tx >> 4;               // 0: lanes 0-15, 1: lanes 16-31
    const int lm   = tx & 15;

    v8f acc;
#pragma unroll
    for (int i = 0; i < 8; ++i) acc[i] = 0.f;

#pragma unroll
    for (int chunk = 0; chunk < NCH; ++chunk) {
        const int kk = chunk * 32;

        if constexpr (K % 32 != 0) {
            // single padded chunk: zero-fill, then pack the valid K/2 pairs
            for (int i = tid; i < 16 * 16; i += 32 * WY)
                As2[i >> 4][i & 15] = 0u;
            for (int i = tid; i < 16 * (K / 2); i += 32 * WY) {
                int r = i / (K / 2), p = i % (K / 2);
                const float2 f = *(const float2*)(A + (size_t)(m0 + r) * K + 2 * p);
                As2[r][p] = pk2(f.x, f.y);
            }
            for (int i = tx; i < 16 * 16; i += 32)
                Bs2[ty][i >> 4][i & 15] = 0u;
            for (int i = tx; i < 16 * (K / 2); i += 32) {
                int kp = i >> 4, c = i & 15;
                float b0 = B[(size_t)(2 * kp)     * Nc + n0 + c];
                float b1 = B[(size_t)(2 * kp + 1) * Nc + n0 + c];
                Bs2[ty][c][kp] = pk2(b0, b1);
            }
        } else {
            // full 32-wide chunk: no guards, wide loads
            for (int i = tid; i < 16 * 16; i += 32 * WY) {
                int r = i >> 4, p = i & 15;
                const float2 f = *(const float2*)(A + (size_t)(m0 + r) * K + kk + 2 * p);
                As2[r][p] = pk2(f.x, f.y);
            }
            for (int i = tx; i < 16 * 16; i += 32) {
                int kp = i >> 4, c = i & 15;
                int k  = kk + 2 * kp;
                float b0 = B[(size_t)k       * Nc + n0 + c];
                float b1 = B[(size_t)(k + 1) * Nc + n0 + c];
                Bs2[ty][c][kp] = pk2(b0, b1);
            }
        }
        __syncthreads();

        // fragments per ISA 7.12.2 wave32 layout, assembled from packed pairs
        union { unsigned u[8]; v16h h; } ua, ub;
#pragma unroll
        for (int v = 0; v < 4; ++v) {
            ua.u[v]     = As2[lm][(half ? 4 : 0) + v];       // K=0..7 / 8..15
            ua.u[v + 4] = As2[lm][(half ? 12 : 8) + v];      // K=16..23 / 24..31
        }
#pragma unroll
        for (int v = 0; v < 8; ++v)
            ub.u[v] = Bs2[ty][lm][((v < 4) ? (half ? 4 : 0) : (half ? 8 : 4)) + v];       // K=0..7,16..23 / 8..15,24..31

        acc = __builtin_amdgcn_wmma_f32_16x16x32_f16(
                  false, ua.h, false, ub.h, (short)0, acc, false, false);
        __syncthreads();
    }

#pragma unroll
    for (int v = 0; v < 8; ++v) {
        int m = m0 + v + half * 8;
        C[(size_t)m * Nc + n0 + lm] = acc[v] + bias[n0 + lm];
    }
    (void)M;
}

// ---------------------------------------------------------------------------
// Edge phase (one thread per (edge, head)); edges E..E+N-1 are self loops.
// C is compile-time (64 / 32) so feature loops vectorize to b128 loads.
// ---------------------------------------------------------------------------
template<int C>
__global__ void edge_score_kernel(const float* __restrict__ xl, const float* __restrict__ xr,
                                  const int* __restrict__ ei, const float* __restrict__ att,
                                  float* __restrict__ score, int* __restrict__ smax,
                                  int E, int N)
{
    int idx = blockIdx.x * blockDim.x + threadIdx.x;
    int total = (E + N) * HEADS;
    if (idx >= total) return;
    int e = idx / HEADS, h = idx % HEADS;
    int src = (e < E) ? ei[e]     : (e - E);
    int dst = (e < E) ? ei[E + e] : (e - E);
    const float4* pl = (const float4*)(xl  + (size_t)src * HEADS * C + h * C);
    const float4* pr = (const float4*)(xr  + (size_t)dst * HEADS * C + h * C);
    const float4* pa = (const float4*)(att + h * C);
    float s = 0.f;
#pragma unroll
    for (int c = 0; c < C / 4; ++c) {
        float4 a = pl[c], b = pr[c], w = pa[c];
        float v0 = a.x + b.x, v1 = a.y + b.y, v2 = a.z + b.z, v3 = a.w + b.w;
        v0 = (v0 > 0.f) ? v0 : NEG_SLOPE * v0;
        v1 = (v1 > 0.f) ? v1 : NEG_SLOPE * v1;
        v2 = (v2 > 0.f) ? v2 : NEG_SLOPE * v2;
        v3 = (v3 > 0.f) ? v3 : NEG_SLOPE * v3;
        s = fmaf(w.x, v0, s); s = fmaf(w.y, v1, s);
        s = fmaf(w.z, v2, s); s = fmaf(w.w, v3, s);
    }
    score[idx] = s;
    atomicMax(&smax[dst * HEADS + h], f2ord(s));
}

__global__ void edge_exp_kernel(float* __restrict__ score, const int* __restrict__ smax,
                                float* __restrict__ den, const int* __restrict__ ei,
                                int E, int N)
{
    int idx = blockIdx.x * blockDim.x + threadIdx.x;
    int total = (E + N) * HEADS;
    if (idx >= total) return;
    int e = idx / HEADS, h = idx % HEADS;
    int dst = (e < E) ? ei[E + e] : (e - E);
    float m  = ord2f(smax[dst * HEADS + h]);
    float ex = __expf(score[idx] - m);
    score[idx] = ex;
    atomicAdd(&den[dst * HEADS + h], ex);
}

template<int C>
__global__ void edge_agg_kernel(const float* __restrict__ ex, const float* __restrict__ den,
                                const float* __restrict__ xl, const int* __restrict__ ei,
                                float* __restrict__ agg, int E, int N)
{
    int idx = blockIdx.x * blockDim.x + threadIdx.x;
    int total = (E + N) * HEADS;
    if (idx >= total) return;
    int e = idx / HEADS, h = idx % HEADS;
    int src = (e < E) ? ei[e]     : (e - E);
    int dst = (e < E) ? ei[E + e] : (e - E);
    float alpha = ex[idx] / den[dst * HEADS + h];
    const float4* pl = (const float4*)(xl + (size_t)src * HEADS * C + h * C);
    float* pd = agg + (size_t)dst * HEADS * C + h * C;
#pragma unroll
    for (int c = 0; c < C / 4; ++c) {
        float4 a = pl[c];
        atomicAdd(&pd[4 * c + 0], alpha * a.x);
        atomicAdd(&pd[4 * c + 1], alpha * a.y);
        atomicAdd(&pd[4 * c + 2], alpha * a.z);
        atomicAdd(&pd[4 * c + 3], alpha * a.w);
    }
}

// ---------------------------------------------------------------------------
// head mean + bias -> LayerNorm -> ReLU ; one wave32 per node, D <= 64
// ---------------------------------------------------------------------------
__global__ void mean_ln_relu_kernel(const float* __restrict__ agg, const float* __restrict__ bias,
                                    const float* __restrict__ g, const float* __restrict__ b,
                                    float* __restrict__ out, int N, int D)
{
    int node = blockIdx.x * blockDim.y + threadIdx.y;
    if (node >= N) return;
    int lane = threadIdx.x;              // wave32
    const float* p = agg + (size_t)node * HEADS * D;

    int d0 = lane, d1 = lane + 32;
    float v0 = 0.f, v1 = 0.f;
    if (d0 < D) {
        float s = 0.f;
        for (int h = 0; h < HEADS; ++h) s += p[h * D + d0];
        v0 = s * (1.f / HEADS) + bias[d0];
    }
    if (d1 < D) {
        float s = 0.f;
        for (int h = 0; h < HEADS; ++h) s += p[h * D + d1];
        v1 = s * (1.f / HEADS) + bias[d1];
    }
    float sum = (d0 < D ? v0 : 0.f) + (d1 < D ? v1 : 0.f);
#pragma unroll
    for (int o = 16; o > 0; o >>= 1) sum += __shfl_xor(sum, o, 32);
    float mu = sum / (float)D;
    float q0 = (d0 < D) ? (v0 - mu) : 0.f;
    float q1 = (d1 < D) ? (v1 - mu) : 0.f;
    float vs = q0 * q0 + q1 * q1;
#pragma unroll
    for (int o = 16; o > 0; o >>= 1) vs += __shfl_xor(vs, o, 32);
    float inv = rsqrtf(vs / (float)D + LN_EPS);
    if (d0 < D) {
        float y = q0 * inv * g[d0] + b[d0];
        out[(size_t)node * D + d0] = (y > 0.f) ? y : 0.f;
    }
    if (d1 < D) {
        float y = q1 * inv * g[d1] + b[d1];
        out[(size_t)node * D + d1] = (y > 0.f) ? y : 0.f;
    }
}

// ---------------------------------------------------------------------------
// per-graph mean pool
// ---------------------------------------------------------------------------
__global__ void pool_accum_kernel(const float* __restrict__ h, const int* __restrict__ batch,
                                  float* __restrict__ pooled, float* __restrict__ cnt,
                                  int N, int D)
{
    int n = blockIdx.x * blockDim.x + threadIdx.x;
    if (n >= N) return;
    int gi = batch[n];
    for (int d = 0; d < D; ++d)
        atomicAdd(&pooled[(size_t)gi * D + d], h[(size_t)n * D + d]);
    atomicAdd(&cnt[gi], 1.f);
}

__global__ void pool_final_kernel(const float* __restrict__ pooled, const float* __restrict__ cnt,
                                  float* __restrict__ out, int G, int D)
{
    int i = blockIdx.x * blockDim.x + threadIdx.x;
    if (i >= G * D) return;
    out[i] = pooled[i] / fmaxf(cnt[i / D], 1.f);
}

__global__ void fill_int_kernel(int* __restrict__ p, int v, int n)
{
    int i = blockIdx.x * blockDim.x + threadIdx.x;
    if (i < n) p[i] = v;
}

// ---------------------------------------------------------------------------
extern "C" void kernel_launch(void* const* d_in, const int* in_sizes, int n_in,
                              void* d_out, int out_size, void* d_ws, size_t ws_size,
                              hipStream_t stream)
{
    const float* x    = (const float*)d_in[0];
    const int*   ei   = (const int*)  d_in[1];
    const int*   batch= (const int*)  d_in[2];
    const float* Wl1  = (const float*)d_in[3];
    const float* bl1  = (const float*)d_in[4];
    const float* Wr1  = (const float*)d_in[5];
    const float* br1  = (const float*)d_in[6];
    const float* att1 = (const float*)d_in[7];
    const float* b1   = (const float*)d_in[8];
    const float* g1   = (const float*)d_in[9];
    const float* be1  = (const float*)d_in[10];
    const float* Wl2  = (const float*)d_in[11];
    const float* bl2  = (const float*)d_in[12];
    const float* Wr2  = (const float*)d_in[13];
    const float* br2  = (const float*)d_in[14];
    const float* att2 = (const float*)d_in[15];
    const float* b2   = (const float*)d_in[16];
    const float* g2   = (const float*)d_in[17];
    const float* be2  = (const float*)d_in[18];

    const int IN = 8, HID = 64, OUT = 32;
    const int N  = in_sizes[0] / IN;     // 20000, multiple of 16
    const int E  = in_sizes[1] / 2;
    const int G  = out_size / OUT;
    const int Et = E + N;

    // workspace layout (floats), buffers reused across layers
    float* ws = (float*)d_ws;
    size_t off = 0;
    float* xl    = ws + off; off += (size_t)N * HEADS * HID;   // layer1 xl / layer2 xl
    float* xr    = ws + off; off += (size_t)N * HEADS * HID;   // layer1 xr / layer2 xr
    float* agg   = ws + off; off += (size_t)N * HEADS * HID;   // attention aggregate
    float* score = ws + off; off += (size_t)Et * HEADS;        // scores -> exp
    float* den   = ws + off; off += (size_t)N * HEADS;
    int*   smax  = (int*)(ws + off); off += (size_t)N * HEADS;
    float* h1    = ws + off; off += (size_t)N * HID;
    float* h2    = ws + off; off += (size_t)N * OUT;
    float* pooled= ws + off; off += (size_t)G * OUT;
    float* cnt   = ws + off; off += (size_t)G;

    const dim3 blkG(32, 4);
    const int  nthE = Et * HEADS;
    const int  ebl  = 256;
    float* outp = (float*)d_out;

    // ---------------- layer 1 (K=8, Nc=256) ----------------
    gemm_wmma<4, 8><<<dim3(N / 16, (HEADS * HID) / 64), blkG, 0, stream>>>(
        x, Wl1, bl1, xl, N, HEADS * HID);
    gemm_wmma<4, 8><<<dim3(N / 16, (HEADS * HID) / 64), blkG, 0, stream>>>(
        x, Wr1, br1, xr, N, HEADS * HID);

    (void)hipMemsetAsync(den, 0, (size_t)N * HEADS * sizeof(float), stream);
    (void)hipMemsetAsync(agg, 0, (size_t)N * HEADS * HID * sizeof(float), stream);
    fill_int_kernel<<<(N * HEADS + 255) / 256, 256, 0, stream>>>(smax, (int)0x80000000, N * HEADS);

    edge_score_kernel<64><<<(nthE + ebl - 1) / ebl, ebl, 0, stream>>>(xl, xr, ei, att1, score, smax, E, N);
    edge_exp_kernel      <<<(nthE + ebl - 1) / ebl, ebl, 0, stream>>>(score, smax, den, ei, E, N);
    edge_agg_kernel<64>  <<<(nthE + ebl - 1) / ebl, ebl, 0, stream>>>(score, den, xl, ei, agg, E, N);

    mean_ln_relu_kernel<<<dim3((N + 3) / 4), dim3(32, 4), 0, stream>>>(agg, b1, g1, be1, h1, N, HID);

    // ---------------- layer 2 (K=64, Nc=128) ----------------
    gemm_wmma<4, 64><<<dim3(N / 16, (HEADS * OUT) / 64), blkG, 0, stream>>>(
        h1, Wl2, bl2, xl, N, HEADS * OUT);
    gemm_wmma<4, 64><<<dim3(N / 16, (HEADS * OUT) / 64), blkG, 0, stream>>>(
        h1, Wr2, br2, xr, N, HEADS * OUT);

    (void)hipMemsetAsync(den, 0, (size_t)N * HEADS * sizeof(float), stream);
    (void)hipMemsetAsync(agg, 0, (size_t)N * HEADS * OUT * sizeof(float), stream);
    fill_int_kernel<<<(N * HEADS + 255) / 256, 256, 0, stream>>>(smax, (int)0x80000000, N * HEADS);

    edge_score_kernel<32><<<(nthE + ebl - 1) / ebl, ebl, 0, stream>>>(xl, xr, ei, att2, score, smax, E, N);
    edge_exp_kernel      <<<(nthE + ebl - 1) / ebl, ebl, 0, stream>>>(score, smax, den, ei, E, N);
    edge_agg_kernel<32>  <<<(nthE + ebl - 1) / ebl, ebl, 0, stream>>>(score, den, xl, ei, agg, E, N);

    mean_ln_relu_kernel<<<dim3((N + 3) / 4), dim3(32, 4), 0, stream>>>(agg, b2, g2, be2, h2, N, OUT);

    // ---------------- pooling ----------------
    (void)hipMemsetAsync(pooled, 0, (size_t)G * OUT * sizeof(float), stream);
    (void)hipMemsetAsync(cnt,    0, (size_t)G * sizeof(float), stream);
    pool_accum_kernel<<<(N + 255) / 256, 256, 0, stream>>>(h2, batch, pooled, cnt, N, OUT);
    pool_final_kernel<<<(G * OUT + 255) / 256, 256, 0, stream>>>(pooled, cnt, outp, G, OUT);

    (void)n_in; (void)ws_size;
}
